// ScaledDotProductAttention_19421842112844
// MI455X (gfx1250) — matrix operand order repaired and checked by reading, not compiled, not run on hardware
//
#include <hip/hip_runtime.h>
#include <hip/hip_bf16.h>
#include <math.h>

typedef __attribute__((ext_vector_type(16))) _Float16 v16h;
typedef __attribute__((ext_vector_type(8)))  float    v8f;

#define MODEL_DIM 1024
#define HEAD_DIM  128
#define BATCH     4
#define SEQ       4096

#define K_STRIDE  136   // 32x128 K tile, padded row stride (halves); 272B rows -> conflict-free B-frag reads
#define V_STRIDE  40    // 128x32 V^T tile, padded row stride (halves); 80B rows  -> conflict-free B-frag reads

// ---------------------------------------------------------------------------
// Projection GEMM: Y[r, n] = sum_k X[r, k] * W[n, k]   (Y f16, X/W f32)
// 256 threads = 8 waves; each wave computes one 16x16 tile of a 16-row strip.
// ---------------------------------------------------------------------------
__global__ __launch_bounds__(256) void proj_kernel(const float* __restrict__ X,
                                                   const float* __restrict__ W,
                                                   _Float16* __restrict__ Y) {
  const int lane = threadIdx.x & 31;
  const int wave = threadIdx.x >> 5;      // 0..7 -> output col tile
  const int row0 = blockIdx.x * 16;       // 16 rows of [B*S]
  const int n0   = wave * 16;
  const int m    = lane & 15;
  const int hi   = lane >> 4;

  v8f acc = {};
  for (int k0 = 0; k0 < MODEL_DIM; k0 += 32) {
    v16h a, b;
    {   // A fragment: lane row m, K = k0+(hi?16:0)..+15 (matches B)
      const float* p = X + (size_t)(row0 + m) * MODEL_DIM + k0 + (hi ? 16 : 0);
#pragma unroll
      for (int i = 0; i < 16; ++i) a[i] = (_Float16)p[i];
    }
    {   // B fragment: B[k][n] = W[n0+n][k]; lane col n=m, K = k0+(hi?16:0)..+15
      const float* p = W + (size_t)(n0 + m) * MODEL_DIM + k0 + (hi ? 16 : 0);
#pragma unroll
      for (int i = 0; i < 16; ++i) b[i] = (_Float16)p[i];
    }
    acc = __builtin_amdgcn_wmma_f32_16x16x32_f16(false, a, false, b,
                                                 (short)0, acc, false, false);
  }
  _Float16* o = Y + (size_t)(row0 + hi * 8) * HEAD_DIM + n0 + m;
#pragma unroll
  for (int r = 0; r < 8; ++r) o[(size_t)r * HEAD_DIM] = (_Float16)acc[r];
}

// ---------------------------------------------------------------------------
// Flash attention over projected f16 q/k/v. 128 threads = 4 waves.
// Wave w owns query rows [blockIdx.x*64 + 16w, +16). Per iteration the block
// stages a 32-key K tile (async global->LDS, padded) and a V^T tile, then
// each wave does 8 WMMAs for S = Q K^T (two 16-col tiles) and 8 full-K=32
// WMMAs for O += P V.
// ---------------------------------------------------------------------------
__global__ __launch_bounds__(128) void fattn_kernel(const _Float16* __restrict__ Qp,
                                                    const _Float16* __restrict__ Kp,
                                                    const _Float16* __restrict__ Vp,
                                                    const int* __restrict__ causal_flag,
                                                    float* __restrict__ Out) {
  __shared__ __align__(16) _Float16 k_lds[32 * K_STRIDE];        // K tile rows j0..j0+31
  __shared__ __align__(16) _Float16 vT_lds[HEAD_DIM * V_STRIDE]; // vT[d][kk] = V[j0+kk][d]
  __shared__ __align__(16) _Float16 p_lds[4][16 * V_STRIDE];     // per-wave P (16x32, padded)

  const int lane = threadIdx.x & 31;
  const int wave = threadIdx.x >> 5;               // 0..3
  const int b    = blockIdx.y;
  const int r0b  = blockIdx.x * 64;
  const int r0   = r0b + wave * 16;
  const int m    = lane & 15;
  const int hi   = lane >> 4;
  const bool causal = (*causal_flag) != 0;

  const size_t base = (size_t)b * SEQ * HEAD_DIM;

  // Q tile in registers: 4 A-fragments covering d = 0..127
  v16h qf[4];
#pragma unroll
  for (int c = 0; c < 4; ++c) {
    const _Float16* p = Qp + base + (size_t)(r0 + m) * HEAD_DIM + c * 32 + (hi ? 16 : 0);
#pragma unroll
    for (int i = 0; i < 16; ++i) qf[c][i] = p[i];
  }

  v8f o[8];
#pragma unroll
  for (int t = 0; t < 8; ++t) o[t] = (v8f){};
  float mrow[8], lrow[8];
#pragma unroll
  for (int r = 0; r < 8; ++r) { mrow[r] = -3.0e38f; lrow[r] = 0.0f; }

  const int   jend  = causal ? (r0b + 64) : SEQ;   // uniform per block, mult of 32
  const int   qlast = r0 + 15;                     // uniform per wave
  const float scale = 0.08838834764831845f;        // 1/sqrt(128)

  for (int j0 = 0; j0 < jend; j0 += 32) {
    __syncthreads();
    {   // ---- stage K tile via async global->LDS (512 x 16B chunks, padded rows) ----
      const int t = threadIdx.x;
#pragma unroll
      for (int c = 0; c < 4; ++c) {
        const int idx = c * 128 + t;               // chunk 0..511
        const int kr  = idx >> 4;                  // key row 0..31
        const int c8  = idx & 15;                  // 8-half column chunk
        const unsigned la =
            (unsigned)(uintptr_t)(const void*)&k_lds[kr * K_STRIDE + c8 * 8];
        const unsigned long long ga = (unsigned long long)(uintptr_t)(const void*)
            (Kp + base + (size_t)(j0 + kr) * HEAD_DIM + c8 * 8);
        asm volatile("global_load_async_to_lds_b128 %0, %1, off"
                     :: "v"(la), "v"(ga) : "memory");
      }
      // ---- stage V^T (transpose in VALU; padded stride) ----
      const int kk = t & 31;                       // key row 0..31
      const int d0 = (t >> 5) * 32;                // d chunk per wave
      const _Float16* vs = Vp + base + (size_t)(j0 + kk) * HEAD_DIM + d0;
#pragma unroll
      for (int i = 0; i < 32; ++i) vT_lds[(d0 + i) * V_STRIDE + kk] = vs[i];
      // prefetch next K/V tiles while this tile is consumed
      if (j0 + 32 < jend) {
        __builtin_prefetch(Kp + base + (size_t)(j0 + 32 + (t >> 2)) * HEAD_DIM, 0, 1);
        __builtin_prefetch(Vp + base + (size_t)(j0 + 32 + (t >> 2)) * HEAD_DIM, 0, 1);
      }
      asm volatile("s_wait_asynccnt 0" ::: "memory");
    }
    __syncthreads();

    if (causal && j0 > qlast) continue;            // wave past its causal range

    // ---- S = Q K^T : two 16x16 tiles (keys j0+h*16 .. +15) ----
    v8f s0 = {}, s1 = {};
#pragma unroll
    for (int c = 0; c < 4; ++c) {
      v16h bf0, bf1;
      const _Float16* p0 = k_lds + m * K_STRIDE + c * 32 + (hi ? 16 : 0);
      const _Float16* p1 = p0 + 16 * K_STRIDE;
#pragma unroll
      for (int i = 0; i < 16; ++i) { bf0[i] = p0[i]; bf1[i] = p1[i]; }
      s0 = __builtin_amdgcn_wmma_f32_16x16x32_f16(false, qf[c], false, bf0,
                                                  (short)0, s0, false, false);
      s1 = __builtin_amdgcn_wmma_f32_16x16x32_f16(false, qf[c], false, bf1,
                                                  (short)0, s1, false, false);
    }

    // ---- causal mask + online softmax over 32 columns ----
#pragma unroll
    for (int r = 0; r < 8; ++r) {
      const int qrow = r0 + hi * 8 + r;
      float x0 = s0[r] * scale;
      float x1 = s1[r] * scale;
      if (causal && (j0 + m)      > qrow) x0 = -3.0e38f;
      if (causal && (j0 + 16 + m) > qrow) x1 = -3.0e38f;
      float rm = fmaxf(x0, x1);
#pragma unroll
      for (int off = 1; off < 16; off <<= 1) rm = fmaxf(rm, __shfl_xor(rm, off, 32));
      const float mnew  = fmaxf(mrow[r], rm);
      const float alpha = __expf(mrow[r] - mnew);
      const float p0    = __expf(x0 - mnew);
      const float p1    = __expf(x1 - mnew);
      float rs = p0 + p1;
#pragma unroll
      for (int off = 1; off < 16; off <<= 1) rs += __shfl_xor(rs, off, 32);
      lrow[r] = lrow[r] * alpha + rs;
      mrow[r] = mnew;
#pragma unroll
      for (int t = 0; t < 8; ++t) o[t][r] *= alpha;
      p_lds[wave][(hi * 8 + r) * V_STRIDE + m]      = (_Float16)p0;
      p_lds[wave][(hi * 8 + r) * V_STRIDE + 16 + m] = (_Float16)p1;
    }

    // ---- P: C-layout -> full 16x32 A-fragment via per-wave LDS ----
    asm volatile("s_wait_dscnt 0" ::: "memory");
    v16h pa;
    {
      const _Float16* pp = p_lds[wave] + m * V_STRIDE + (hi ? 16 : 0);
#pragma unroll
      for (int i = 0; i < 16; ++i) pa[i] = pp[i];         // K = (hi?16:0)+0..15
    }

    // ---- O += P V : 8 column tiles of d, full K=32 ----
#pragma unroll
    for (int t = 0; t < 8; ++t) {
      v16h bv;   // B[k][n] = V[j0+k][t*16+n] = vT[(t*16+n)*V_STRIDE + k]
      const _Float16* vv = vT_lds + (t * 16 + m) * V_STRIDE + (hi ? 16 : 0);
#pragma unroll
      for (int i = 0; i < 16; ++i) bv[i] = vv[i];
      o[t] = __builtin_amdgcn_wmma_f32_16x16x32_f16(false, pa, false, bv,
                                                    (short)0, o[t], false, false);
    }
  }

  // ---- epilogue: O / l -> Out [B,S,128] f32 ----
  float* op = Out + base + (size_t)(r0 + hi * 8) * HEAD_DIM + m;
#pragma unroll
  for (int t = 0; t < 8; ++t)
#pragma unroll
    for (int r = 0; r < 8; ++r)
      op[(size_t)r * HEAD_DIM + t * 16] = o[t][r] / lrow[r];
}

extern "C" void kernel_launch(void* const* d_in, const int* in_sizes, int n_in,
                              void* d_out, int out_size, void* d_ws, size_t ws_size,
                              hipStream_t stream) {
  (void)in_sizes; (void)n_in; (void)out_size; (void)ws_size;
  const float* query = (const float*)d_in[0];
  const float* key   = (const float*)d_in[1];
  const float* value = (const float*)d_in[2];
  const float* Wq    = (const float*)d_in[3];
  const float* Wk    = (const float*)d_in[4];
  const float* Wv    = (const float*)d_in[5];
  const int* is_causal = (const int*)d_in[6];

  _Float16* qp = (_Float16*)d_ws;                       // 4 MB each
  _Float16* kp = qp + (size_t)BATCH * SEQ * HEAD_DIM;
  _Float16* vp = kp + (size_t)BATCH * SEQ * HEAD_DIM;

  dim3 pgrid(BATCH * SEQ / 16);
  proj_kernel<<<pgrid, 256, 0, stream>>>(query, Wq, qp);
  proj_kernel<<<pgrid, 256, 0, stream>>>(key,   Wk, kp);
  proj_kernel<<<pgrid, 256, 0, stream>>>(value, Wv, vp);

  dim3 agrid(SEQ / 64, BATCH);
  fattn_kernel<<<agrid, 128, 0, stream>>>(qp, kp, vp, is_causal, (float*)d_out);
}
